// HMP_Equiformer_Net_Ablation_77017353552150
// MI455X (gfx1250) — matrix operand order repaired and checked by reading, not compiled, not run on hardware
//
#include <hip/hip_runtime.h>
#include <hip/hip_bf16.h>
#include <math.h>

// ---------------- problem constants (from reference) ----------------
#define NEDGE  100000
#define NNODE  8192
#define NGRAPH 512

#define SQ3f   1.7320508075688772f
#define SQ5f   2.2360679774997896f
#define SQ15f  3.8729833462074170f
#define INV_DEG 0.25342853f   // 1/sqrt(15.57)
#define INV_ANN 0.23550565f   // 1/sqrt(18.03)

typedef __attribute__((ext_vector_type(16))) _Float16 v16h;
typedef __attribute__((ext_vector_type(8)))  float    v8f;

__device__ __forceinline__ float sigf(float x)   { return 1.0f / (1.0f + __expf(-x)); }
__device__ __forceinline__ float siluf(float x)  { return x * sigf(x); }
__device__ __forceinline__ float dsiluf(float x) { float s = sigf(x); return s * (1.0f + x * (1.0f - s)); }

// order-preserving float<->uint encoding for atomic segment-max
__device__ __forceinline__ unsigned fenc(float v) {
  unsigned b = __float_as_uint(v);
  return (b & 0x80000000u) ? ~b : (b | 0x80000000u);
}
__device__ __forceinline__ float fdec(unsigned u) {
  unsigned b = (u & 0x80000000u) ? (u & 0x7fffffffu) : ~u;
  return __uint_as_float(b);
}

// =====================================================================
// Generic f16 WMMA GEMM. One wave computes a 16 x 64 output strip:
//   raw = A[M x K] @ B[K x N]      (A rows optionally gathered via gidx)
//   C   (f32, optional)  = raw  (or += raw if accum)
//   C16 (f16, optional)  = act(raw)   act: 0=id, 1=silu, 2=sigmoid
// M multiple of 16, N multiple of 16, K multiple of 32.
// The A fragment is packed once per K-step and reused by up to 4
// unrolled v_wmma ops (one per 16-wide N sub-tile). Sub-tile guards are
// wave-uniform so EXEC stays all-ones across the WMMAs (ISA 7.12 req).
// =====================================================================
__global__ __launch_bounds__(32)
void k_gemm(const _Float16* __restrict__ A, int lda,
            const int* __restrict__ gidx, int goff,
            const _Float16* __restrict__ B, int ldb,
            float* __restrict__ C, int ldc, int coff, int accum,
            _Float16* __restrict__ C16, int ldc16, int c16off, int act,
            int N, int K)
{
  __shared__ _Float16 As[16 * 32];
  __shared__ _Float16 Bs[32 * 64];

  const int lane = threadIdx.x;
  const int m0 = blockIdx.y * 16;
  const int n0 = blockIdx.x * 64;
  int nt = (N - n0) >> 4;            // 16-wide sub-tiles in this strip
  if (nt > 4) nt = 4;                // wave-uniform

  v8f acc0 = {}, acc1 = {}, acc2 = {}, acc3 = {};

  for (int k0 = 0; k0 < K; k0 += 32) {
    // ---- stage A tile (16 rows x 32 cols), 2 lanes per row ----
    {
      const int row  = lane >> 1;
      const int hb   = (lane & 1) << 4;
      const long rbase = (gidx ? (long)gidx[m0 + row] : (long)(m0 + row)) * (long)lda
                         + (long)goff + k0 + hb;
      const _Float16* ap = A + rbase;
      #pragma unroll
      for (int i = 0; i < 16; ++i) As[row * 32 + hb + i] = ap[i];
      if (k0 + 32 < K) __builtin_prefetch(ap + 32, 0, 0);  // global_prefetch_b8
    }
    // ---- stage B tile (32 rows x up to 64 cols), 1 lane per K-row ----
    {
      const _Float16* bp = B + (long)(k0 + lane) * (long)ldb + n0;
      #pragma unroll
      for (int s = 0; s < 4; ++s) {
        if (s < nt) {
          #pragma unroll
          for (int i = 0; i < 16; ++i) Bs[lane * 64 + s * 16 + i] = bp[s * 16 + i];
        }
      }
      if (k0 + 32 < K) __builtin_prefetch(bp + 32L * ldb, 0, 0);
    }
    __syncthreads();

    // ---- pack A fragment per CDNA5 ISA 7.12.2 (16-bit A 16x32) ----
    v16h af;
    {
      const int arow = lane & 15;
      const int kb   = (lane < 16) ? 0 : 8;
      #pragma unroll
      for (int v = 0; v < 4; ++v) {
        af[2 * v]     = As[arow * 32 + kb + 2 * v];
        af[2 * v + 1] = As[arow * 32 + kb + 2 * v + 1];
        af[2 * v + 8] = As[arow * 32 + kb + 16 + 2 * v];
        af[2 * v + 9] = As[arow * 32 + kb + 16 + 2 * v + 1];
      }
    }
    const int krow = (lane < 16) ? 0 : 8;
    const int ncol = lane & 15;
    v16h bf;
    #pragma unroll
    for (int h = 0; h < 16; ++h) bf[h] = Bs[(krow + h + ((h >> 3) << 3)) * 64 + ncol];
    acc0 = __builtin_amdgcn_wmma_f32_16x16x32_f16(false, af, false, bf, (short)0, acc0, false, false);
    if (nt > 1) {
      #pragma unroll
      for (int h = 0; h < 16; ++h) bf[h] = Bs[(krow + h + ((h >> 3) << 3)) * 64 + 16 + ncol];
      acc1 = __builtin_amdgcn_wmma_f32_16x16x32_f16(false, af, false, bf, (short)0, acc1, false, false);
    }
    if (nt > 2) {
      #pragma unroll
      for (int h = 0; h < 16; ++h) bf[h] = Bs[(krow + h + ((h >> 3) << 3)) * 64 + 32 + ncol];
      acc2 = __builtin_amdgcn_wmma_f32_16x16x32_f16(false, af, false, bf, (short)0, acc2, false, false);
    }
    if (nt > 3) {
      #pragma unroll
      for (int h = 0; h < 16; ++h) bf[h] = Bs[(krow + h + ((h >> 3) << 3)) * 64 + 48 + ncol];
      acc3 = __builtin_amdgcn_wmma_f32_16x16x32_f16(false, af, false, bf, (short)0, acc3, false, false);
    }
    __syncthreads();
  }

  // ---- writeback: VGPR v holds M=v (lanes 0-15) / M=v+8 (lanes 16-31) ----
  const int colL = lane & 15;
  const int rbv  = m0 + ((lane < 16) ? 0 : 8);
  auto writeback = [&](const v8f& a, int s) {
    const int col = n0 + s * 16 + colL;
    #pragma unroll
    for (int v = 0; v < 8; ++v) {
      const int row = rbv + v;
      const float x = a[v];
      if (C) {
        long ci = (long)row * ldc + coff + col;
        if (accum) C[ci] += x; else C[ci] = x;
      }
      if (C16) {
        float y = x;
        if (act == 1) y = siluf(x);
        else if (act == 2) y = sigf(x);
        C16[(long)row * ldc16 + c16off + col] = (_Float16)y;
      }
    }
  };
  writeback(acc0, 0);
  if (nt > 1) writeback(acc1, 1);
  if (nt > 2) writeback(acc2, 2);
  if (nt > 3) writeback(acc3, 3);
}

// =====================================================================
// small utility kernels
// =====================================================================
__global__ void k_zero_f32(float* p, long n) {
  long i = (long)blockIdx.x * blockDim.x + threadIdx.x;
  if (i < n) p[i] = 0.0f;
}
__global__ void k_fill_u32(unsigned* p, unsigned v, long n) {
  long i = (long)blockIdx.x * blockDim.x + threadIdx.x;
  if (i < n) p[i] = v;
}
__global__ void k_cvt16(const float* __restrict__ s, _Float16* __restrict__ d, long n) {
  long i = (long)blockIdx.x * blockDim.x + threadIdx.x;
  if (i < n) d[i] = (_Float16)s[i];
}
__global__ void k_transpose16(const float* __restrict__ s, _Float16* __restrict__ d, int R, int Cc) {
  long i = (long)blockIdx.x * blockDim.x + threadIdx.x;
  if (i < (long)R * Cc) {
    int r = (int)(i / Cc), c = (int)(i % Cc);
    d[(long)c * R + r] = (_Float16)s[i];
  }
}

// =====================================================================
// edge geometry: r, u, sh1, sh2, rbf (f16)
// =====================================================================
__global__ void k_edge_geom(const float* __restrict__ pos,
                            const int* __restrict__ esrc, const int* __restrict__ edst,
                            const float* __restrict__ cent, const float* __restrict__ wid,
                            float* __restrict__ r_, float* __restrict__ u_,
                            float* __restrict__ sh1_, float* __restrict__ sh2_,
                            _Float16* __restrict__ rbfh)
{
  int e = blockIdx.x * blockDim.x + threadIdx.x;
  if (e >= NEDGE) return;
  int s = esrc[e], d = edst[e];
  float vx = pos[(long)s * 3 + 0] - pos[(long)d * 3 + 0];
  float vy = pos[(long)s * 3 + 1] - pos[(long)d * 3 + 1];
  float vz = pos[(long)s * 3 + 2] - pos[(long)d * 3 + 2];
  float r = sqrtf(vx * vx + vy * vy + vz * vz + 1e-12f);
  float x = vx / r, y = vy / r, z = vz / r;
  r_[e] = r;
  u_[(long)e * 3 + 0] = x; u_[(long)e * 3 + 1] = y; u_[(long)e * 3 + 2] = z;
  sh1_[(long)e * 3 + 0] = SQ3f * x;
  sh1_[(long)e * 3 + 1] = SQ3f * y;
  sh1_[(long)e * 3 + 2] = SQ3f * z;
  sh2_[(long)e * 5 + 0] = SQ15f * x * y;
  sh2_[(long)e * 5 + 1] = SQ15f * y * z;
  sh2_[(long)e * 5 + 2] = 0.5f * SQ5f * (3.0f * z * z - 1.0f);
  sh2_[(long)e * 5 + 3] = SQ15f * x * z;
  sh2_[(long)e * 5 + 4] = 0.5f * SQ15f * (x * x - y * y);
  #pragma unroll 4
  for (int k = 0; k < 128; ++k) {
    float t = (r - cent[k]) / wid[k];
    rbfh[(long)e * 128 + k] = (_Float16)__expf(-0.5f * t * t);
  }
}

// node feature init: f0 from atom table, f1/f2 zero. block = node, 192 thr.
__global__ void k_node_init(const int* __restrict__ atoms, const float* __restrict__ table,
                            float* __restrict__ f0, float* __restrict__ f1, float* __restrict__ f2)
{
  int n = blockIdx.x, t = threadIdx.x;
  if (t < 128) {
    int a = atoms[n], m;
    switch (a) { case 1: m = 0; break; case 6: m = 1; break; case 7: m = 2; break;
                 case 8: m = 3; break; case 9: m = 4; break; default: m = 0; }
    f0[(long)n * 128 + t] = table[(long)m * 128 + t];
  }
  if (t < 192) f1[(long)n * 192 + t] = 0.0f;
  if (t < 160) f2[(long)n * 160 + t] = 0.0f;
}

// embedding scatter: f += inv * g (x sh). block = edge, 192 thr.
__global__ void k_scatter_embed(const int* __restrict__ edst, const _Float16* __restrict__ g16,
                                const float* __restrict__ sh1, const float* __restrict__ sh2,
                                float* __restrict__ f0, float* __restrict__ f1, float* __restrict__ f2)
{
  int e = blockIdx.x, t = threadIdx.x;
  int d = edst[e];
  long eb = e;
  if (t < 128) atomicAdd(&f0[(long)d * 128 + t], INV_DEG * (float)g16[eb * 224 + t]);
  if (t < 64) {
    float gv = INV_DEG * (float)g16[eb * 224 + 128 + t];
    #pragma unroll
    for (int x = 0; x < 3; ++x)
      atomicAdd(&f1[(long)d * 192 + x * 64 + t], gv * sh1[eb * 3 + x]);
  }
  if (t < 32) {
    float gv = INV_DEG * (float)g16[eb * 224 + 192 + t];
    #pragma unroll
    for (int x = 0; x < 5; ++x)
      atomicAdd(&f2[(long)d * 160 + x * 32 + t], gv * sh2[eb * 5 + x]);
  }
}

// combine edge messages + attention logit. block = edge, 128 thr.
// m0h/m1h/m2h updated in place: m = (s@Wv)*g + (s0@Ws)*sh
__global__ void k_combine(const _Float16* __restrict__ g16,
                          _Float16* __restrict__ m0h, const _Float16* __restrict__ t0s,
                          const _Float16* __restrict__ ts1, const _Float16* __restrict__ ts2,
                          _Float16* __restrict__ m1h, _Float16* __restrict__ m2h,
                          const _Float16* __restrict__ za, const _Float16* __restrict__ zb,
                          const _Float16* __restrict__ ze, const float* __restrict__ wa,
                          const float* __restrict__ sh1, const float* __restrict__ sh2,
                          float* __restrict__ logit)
{
  __shared__ float zr[32];
  int e = blockIdx.x, t = threadIdx.x;
  long eb = e;
  if (t < 128) {
    float m = (float)m0h[eb * 128 + t] * (float)g16[eb * 224 + t] + (float)t0s[eb * 128 + t];
    m0h[eb * 128 + t] = (_Float16)m;
  }
  if (t < 64) {
    float g1 = (float)g16[eb * 224 + 128 + t];
    float s1 = (float)ts1[eb * 64 + t];
    #pragma unroll
    for (int x = 0; x < 3; ++x) {
      long ix = eb * 192 + x * 64 + t;
      m1h[ix] = (_Float16)((float)m1h[ix] * g1 + s1 * sh1[eb * 3 + x]);
    }
  }
  if (t < 32) {
    float g2 = (float)g16[eb * 224 + 192 + t];
    float s2 = (float)ts2[eb * 32 + t];
    #pragma unroll
    for (int x = 0; x < 5; ++x) {
      long ix = eb * 160 + x * 32 + t;
      m2h[ix] = (_Float16)((float)m2h[ix] * g2 + s2 * sh2[eb * 5 + x]);
    }
    float zz = tanhf((float)za[eb * 32 + t] + (float)zb[eb * 32 + t] + (float)ze[eb * 32 + t]);
    zr[t] = zz * wa[t];
  }
  __syncthreads();
  if (t == 0) {
    float s = 0.0f;
    #pragma unroll
    for (int k = 0; k < 32; ++k) s += zr[k];
    logit[e] = s;
  }
}

__global__ void k_segmax(const int* __restrict__ edst, const float* __restrict__ logit,
                         unsigned* __restrict__ nmax)
{
  int e = blockIdx.x * blockDim.x + threadIdx.x;
  if (e >= NEDGE) return;
  atomicMax(&nmax[edst[e]], fenc(logit[e]));
}
__global__ void k_expsum(const int* __restrict__ edst, const float* __restrict__ logit,
                         const unsigned* __restrict__ nmax,
                         float* __restrict__ ew, float* __restrict__ nsum)
{
  int e = blockIdx.x * blockDim.x + threadIdx.x;
  if (e >= NEDGE) return;
  int d = edst[e];
  float v = __expf(logit[e] - fdec(nmax[d]));
  ew[e] = v;
  atomicAdd(&nsum[d], v);
}

// attention-weighted scatter. block = edge, 192 thr.
__global__ void k_scatter_attn(const int* __restrict__ edst, const float* __restrict__ ew,
                               const float* __restrict__ nsum,
                               const _Float16* __restrict__ m0h, const _Float16* __restrict__ m1h,
                               const _Float16* __restrict__ m2h,
                               float* __restrict__ f0, float* __restrict__ f1, float* __restrict__ f2)
{
  int e = blockIdx.x, t = threadIdx.x;
  int d = edst[e];
  float a = ew[e] / (nsum[d] + 1e-9f);
  long eb = e;
  if (t < 128) atomicAdd(&f0[(long)d * 128 + t], a * (float)m0h[eb * 128 + t]);
  if (t < 192) atomicAdd(&f1[(long)d * 192 + t], a * (float)m1h[eb * 192 + t]);
  if (t < 160) atomicAdd(&f2[(long)d * 160 + t], a * (float)m2h[eb * 160 + t]);
}

// gated channel-mix residual for f1/f2. block = node, 192 thr.
__global__ void k_gatemul(float* __restrict__ f1, float* __restrict__ f2,
                          const _Float16* __restrict__ t1nh, const _Float16* __restrict__ t2nh,
                          const _Float16* __restrict__ g1h, const _Float16* __restrict__ g2h)
{
  int n = blockIdx.x, t = threadIdx.x;
  if (t < 192) f1[(long)n * 192 + t] += (float)t1nh[(long)n * 192 + t] * (float)g1h[(long)n * 64 + (t & 63)];
  if (t < 160) f2[(long)n * 160 + t] += (float)t2nh[(long)n * 160 + t] * (float)g2h[(long)n * 32 + (t & 31)];
}

// LayerNorm over 448 dims (f1 dims interleave as 128+c*3+x, f2 as 320+c*5+x)
__global__ void k_layernorm(const float* __restrict__ f0, const float* __restrict__ f1,
                            const float* __restrict__ f2,
                            const float* __restrict__ lng, const float* __restrict__ lnb,
                            _Float16* __restrict__ ln0h, float* __restrict__ stats)
{
  int n = blockIdx.x * blockDim.x + threadIdx.x;
  if (n >= NNODE) return;
  float s = 0.0f, s2 = 0.0f;
  for (int c = 0; c < 128; ++c) { float v = f0[(long)n * 128 + c]; s += v; s2 += v * v; }
  for (int c = 0; c < 192; ++c) { float v = f1[(long)n * 192 + c]; s += v; s2 += v * v; }
  for (int c = 0; c < 160; ++c) { float v = f2[(long)n * 160 + c]; s += v; s2 += v * v; }
  float mu = s / 448.0f;
  float var = s2 / 448.0f - mu * mu;
  float rs = rsqrtf(var + 1e-5f);
  stats[(long)n * 2 + 0] = mu;
  stats[(long)n * 2 + 1] = rs;
  for (int c = 0; c < 128; ++c) {
    float v = (f0[(long)n * 128 + c] - mu) * rs * lng[c] + lnb[c];
    ln0h[(long)n * 128 + c] = (_Float16)v;
  }
}

__global__ void k_energy(const _Float16* __restrict__ hhh, const float* __restrict__ ho,
                         const int* __restrict__ batch, float* __restrict__ eout)
{
  int n = blockIdx.x * blockDim.x + threadIdx.x;
  if (n >= NNODE) return;
  float e = 0.0f;
  for (int c = 0; c < 128; ++c) e += (float)hhh[(long)n * 128 + c] * ho[c];
  atomicAdd(&eout[batch[n]], e * INV_ANN);
}

// ----- backward pieces -----
__global__ void k_head_bwd(const float* __restrict__ phead, const float* __restrict__ ho,
                           _Float16* __restrict__ dp16)
{
  long i = (long)blockIdx.x * blockDim.x + threadIdx.x;
  if (i >= (long)NNODE * 128) return;
  int c = (int)(i & 127);
  dp16[i] = (_Float16)(INV_ANN * ho[c] * dsiluf(phead[i]));
}

__global__ void k_ln_bwd(const _Float16* __restrict__ dln0h, const float* __restrict__ lng,
                         const float* __restrict__ f0, const float* __restrict__ f1,
                         const float* __restrict__ f2, const float* __restrict__ stats,
                         float* __restrict__ wf0, float* __restrict__ wf1, float* __restrict__ wf2)
{
  int n = blockIdx.x * blockDim.x + threadIdx.x;
  if (n >= NNODE) return;
  float mu = stats[(long)n * 2 + 0], rs = stats[(long)n * 2 + 1];
  float ga = 0.0f, gb = 0.0f;
  for (int c = 0; c < 128; ++c) {
    float dy = (float)dln0h[(long)n * 128 + c] * lng[c];
    float xh = (f0[(long)n * 128 + c] - mu) * rs;
    ga += dy; gb += dy * xh;
  }
  ga /= 448.0f; gb /= 448.0f;
  for (int c = 0; c < 128; ++c) {
    float dy = (float)dln0h[(long)n * 128 + c] * lng[c];
    float xh = (f0[(long)n * 128 + c] - mu) * rs;
    wf0[(long)n * 128 + c] = rs * (dy - ga - xh * gb);
  }
  for (int s = 0; s < 192; ++s) {
    float xh = (f1[(long)n * 192 + s] - mu) * rs;
    wf1[(long)n * 192 + s] = rs * (-ga - xh * gb);
  }
  for (int s = 0; s < 160; ++s) {
    float xh = (f2[(long)n * 160 + s] - mu) * rs;
    wf2[(long)n * 160 + s] = rs * (-ga - xh * gb);
  }
}

// dg (E x 224) from node adjoints through embedding scatter. block = edge, 128 thr.
__global__ void k_emb_bwd_dg(const int* __restrict__ edst,
                             const float* __restrict__ wf0, const float* __restrict__ wf1,
                             const float* __restrict__ wf2,
                             const float* __restrict__ sh1, const float* __restrict__ sh2,
                             _Float16* __restrict__ dgh)
{
  int e = blockIdx.x, t = threadIdx.x;
  int d = edst[e];
  long eb = e;
  if (t < 128) dgh[eb * 224 + t] = (_Float16)(INV_DEG * wf0[(long)d * 128 + t]);
  if (t < 64) {
    float s = 0.0f;
    #pragma unroll
    for (int x = 0; x < 3; ++x) s += wf1[(long)d * 192 + x * 64 + t] * sh1[eb * 3 + x];
    dgh[eb * 224 + 128 + t] = (_Float16)(INV_DEG * s);
  }
  if (t < 32) {
    float s = 0.0f;
    #pragma unroll
    for (int x = 0; x < 5; ++x) s += wf2[(long)d * 160 + x * 32 + t] * sh2[eb * 5 + x];
    dgh[eb * 224 + 192 + t] = (_Float16)(INV_DEG * s);
  }
}

// dsh from node adjoints x embedding g. one thread / edge.
__global__ void k_emb_bwd_sh(const int* __restrict__ edst,
                             const float* __restrict__ wf1, const float* __restrict__ wf2,
                             const _Float16* __restrict__ g16,
                             float* __restrict__ dsh1, float* __restrict__ dsh2)
{
  int e = blockIdx.x * blockDim.x + threadIdx.x;
  if (e >= NEDGE) return;
  int d = edst[e];
  long eb = e;
  for (int x = 0; x < 3; ++x) {
    float s = 0.0f;
    for (int c = 0; c < 64; ++c)
      s += wf1[(long)d * 192 + x * 64 + c] * (float)g16[eb * 224 + 128 + c];
    dsh1[eb * 3 + x] = INV_DEG * s;
  }
  for (int x = 0; x < 5; ++x) {
    float s = 0.0f;
    for (int c = 0; c < 32; ++c)
      s += wf2[(long)d * 160 + x * 32 + c] * (float)g16[eb * 224 + 192 + c];
    dsh2[eb * 5 + x] = INV_DEG * s;
  }
}

__global__ void k_silu_bwd(const _Float16* __restrict__ dy, const float* __restrict__ pre,
                           _Float16* __restrict__ out, long n)
{
  long i = (long)blockIdx.x * blockDim.x + threadIdx.x;
  if (i < n) out[i] = (_Float16)((float)dy[i] * dsiluf(pre[i]));
}

// force accumulation from drbf / dsh. one thread / edge.
__global__ void k_force(const int* __restrict__ esrc, const int* __restrict__ edst,
                        const float* __restrict__ r_, const float* __restrict__ u_,
                        const float* __restrict__ cent, const float* __restrict__ wid,
                        const _Float16* __restrict__ rbfh, const _Float16* __restrict__ drbfh,
                        const float* __restrict__ dsh1, const float* __restrict__ dsh2,
                        float* __restrict__ fout)
{
  int e = blockIdx.x * blockDim.x + threadIdx.x;
  if (e >= NEDGE) return;
  long eb = e;
  float r = r_[e];
  float ux = u_[eb * 3 + 0], uy = u_[eb * 3 + 1], uz = u_[eb * 3 + 2];
  float dr = 0.0f;
  for (int k = 0; k < 128; ++k) {
    float t = (r - cent[k]) / wid[k];
    dr += (float)drbfh[eb * 128 + k] * (float)rbfh[eb * 128 + k] * (-t / wid[k]);
  }
  float dux = SQ3f * dsh1[eb * 3 + 0];
  float duy = SQ3f * dsh1[eb * 3 + 1];
  float duz = SQ3f * dsh1[eb * 3 + 2];
  float d0 = dsh2[eb * 5 + 0], d1 = dsh2[eb * 5 + 1], d2 = dsh2[eb * 5 + 2];
  float d3 = dsh2[eb * 5 + 3], d4 = dsh2[eb * 5 + 4];
  dux += SQ15f * (uy * d0 + uz * d3 + ux * d4);
  duy += SQ15f * (ux * d0 + uz * d1 - uy * d4);
  duz += SQ15f * (uy * d1 + ux * d3) + 3.0f * SQ5f * uz * d2;
  float duu = dux * ux + duy * uy + duz * uz;
  float gx = (dux - ux * duu) / r + ux * dr;
  float gy = (duy - uy * duu) / r + uy * dr;
  float gz = (duz - uz * duu) / r + uz * dr;
  int s = esrc[e], d = edst[e];
  // forces = -grad(pos)
  atomicAdd(&fout[(long)s * 3 + 0], -gx);
  atomicAdd(&fout[(long)s * 3 + 1], -gy);
  atomicAdd(&fout[(long)s * 3 + 2], -gz);
  atomicAdd(&fout[(long)d * 3 + 0],  gx);
  atomicAdd(&fout[(long)d * 3 + 1],  gy);
  atomicAdd(&fout[(long)d * 3 + 2],  gz);
}

// =====================================================================
// host driver
// =====================================================================
extern "C" void kernel_launch(void* const* d_in, const int* in_sizes, int n_in,
                              void* d_out_, int out_size, void* d_ws, size_t ws_size,
                              hipStream_t stream) {
  (void)in_sizes; (void)n_in; (void)out_size; (void)ws_size;
  const float* pos      = (const float*)d_in[0];
  const float* table    = (const float*)d_in[1];
  const float* cent     = (const float*)d_in[2];
  const float* wid      = (const float*)d_in[3];
  const float* deg_W1   = (const float*)d_in[4];
  const float* deg_W2   = (const float*)d_in[5];
  const float* deg_W3   = (const float*)d_in[6];
  const float* bWv0     = (const float*)d_in[7];
  const float* bWv1     = (const float*)d_in[8];
  const float* bWv2     = (const float*)d_in[9];
  const float* bWs0     = (const float*)d_in[10];
  const float* bWs1     = (const float*)d_in[11];
  const float* bWs2     = (const float*)d_in[12];
  const float* brW1     = (const float*)d_in[13];
  const float* brW2     = (const float*)d_in[14];
  const float* brW3     = (const float*)d_in[15];
  const float* bWa_src  = (const float*)d_in[16];
  const float* bWa_dst  = (const float*)d_in[17];
  const float* bWa_e    = (const float*)d_in[18];
  const float* bwa      = (const float*)d_in[19];
  const float* bff1     = (const float*)d_in[20];
  const float* bff2     = (const float*)d_in[21];
  const float* bfg1     = (const float*)d_in[22];
  const float* bfg2     = (const float*)d_in[23];
  const float* bfW1     = (const float*)d_in[24];
  const float* bfW2     = (const float*)d_in[25];
  const float* ln_g     = (const float*)d_in[26];
  const float* ln_b     = (const float*)d_in[27];
  const float* head_W0  = (const float*)d_in[28];
  const float* head_out = (const float*)d_in[29];
  const int*   natom    = (const int*)d_in[30];
  const int*   esrc     = (const int*)d_in[31];
  const int*   edst     = (const int*)d_in[32];
  const int*   batch    = (const int*)d_in[33];

  float* eout = (float*)d_out_;
  float* fout = eout + NGRAPH;

  // ---- bump allocator over workspace (deterministic) ----
  char* wsp = (char*)d_ws;
  auto alloc = [&](size_t bytes) -> void* {
    void* r = (void*)wsp;
    wsp += (bytes + 255) & ~(size_t)255;
    return r;
  };
  auto a16 = [&](long n) { return (_Float16*)alloc((size_t)n * 2); };
  auto a32 = [&](long n) { return (float*)alloc((size_t)n * 4); };

  // f16 weight mirrors
  _Float16* w1h = a16(128 * 64);  _Float16* w2h = a16(64 * 64);  _Float16* w3h = a16(64 * 224);
  _Float16* w1T = a16(64 * 128);  _Float16* w2T = a16(64 * 64);  _Float16* w3T = a16(224 * 64);
  _Float16* hw0h = a16(128 * 128); _Float16* hw0T = a16(128 * 128);
  _Float16* wv0h = a16(8L * 128 * 128); _Float16* wv1h = a16(8L * 64 * 64);  _Float16* wv2h = a16(8L * 32 * 32);
  _Float16* ws0h = a16(8L * 128 * 128); _Float16* ws1h = a16(8L * 128 * 64); _Float16* ws2h = a16(8L * 128 * 32);
  _Float16* rw1h = a16(8L * 128 * 64);  _Float16* rw2h = a16(8L * 64 * 64);  _Float16* rw3h = a16(8L * 64 * 224);
  _Float16* wash = a16(8L * 128 * 32);  _Float16* wadh = a16(8L * 128 * 32); _Float16* waeh = a16(8L * 64 * 32);
  _Float16* ff1h = a16(8L * 128 * 512); _Float16* ff2h = a16(8L * 512 * 128);
  _Float16* fg1h = a16(8L * 128 * 64);  _Float16* fg2h = a16(8L * 128 * 32);
  _Float16* fw1h = a16(8L * 64 * 64);   _Float16* fw2h = a16(8L * 32 * 32);

  // edge buffers
  float* r_   = a32(NEDGE);          float* u_   = a32(3L * NEDGE);
  float* sh1_ = a32(3L * NEDGE);     float* sh2_ = a32(5L * NEDGE);
  float* p1   = a32(64L * NEDGE);    float* p2   = a32(64L * NEDGE);
  float* logit = a32(NEDGE);         float* ew   = a32(NEDGE);
  float* dsh1 = a32(3L * NEDGE);     float* dsh2 = a32(5L * NEDGE);
  _Float16* rbfh = a16(128L * NEDGE);
  _Float16* y1b  = a16(64L * NEDGE); _Float16* y2b = a16(64L * NEDGE);
  _Float16* gEmb = a16(224L * NEDGE); _Float16* gBlk = a16(224L * NEDGE);
  _Float16* t0v  = a16(128L * NEDGE); _Float16* t0s = a16(128L * NEDGE);
  _Float16* ts1  = a16(64L * NEDGE);  _Float16* ts2 = a16(32L * NEDGE);
  _Float16* za   = a16(32L * NEDGE);  _Float16* zb  = a16(32L * NEDGE);  _Float16* ze = a16(32L * NEDGE);
  _Float16* t1m  = a16(192L * NEDGE); _Float16* t2m = a16(160L * NEDGE);

  // node buffers
  float* f0 = a32(128L * NNODE); float* f1 = a32(192L * NNODE); float* f2 = a32(160L * NNODE);
  float* stats = a32(2L * NNODE); float* nsum = a32(NNODE);
  float* phead = a32(128L * NNODE);
  float* wf0 = a32(128L * NNODE); float* wf1 = a32(192L * NNODE); float* wf2 = a32(160L * NNODE);
  unsigned* nmax = (unsigned*)alloc((size_t)NNODE * 4);
  _Float16* f0h = a16(128L * NNODE); _Float16* f1h = a16(192L * NNODE); _Float16* f2h = a16(160L * NNODE);
  _Float16* u1h = a16(512L * NNODE);
  _Float16* g1h = a16(64L * NNODE);  _Float16* g2h = a16(32L * NNODE);
  _Float16* t1nh = a16(192L * NNODE); _Float16* t2nh = a16(160L * NNODE);
  _Float16* ln0h = a16(128L * NNODE); _Float16* hhh = a16(128L * NNODE);
  _Float16* dp16h = a16(128L * NNODE); _Float16* dln0h = a16(128L * NNODE);

  // backward aliases over forward edge temporaries (free after block loop)
  _Float16* dgh   = gBlk;   // E x 224
  _Float16* dy2h  = ts1;    // E x 64
  _Float16* dph   = y1b;    // E x 64
  _Float16* dy1h  = y2b;    // E x 64
  _Float16* dp1h  = t1m;    // E x 64 (uses first 64 of 192)
  _Float16* drbfh = t0s;    // E x 128

  auto cvt = [&](const float* s, _Float16* d, long n) {
    k_cvt16<<<(unsigned)((n + 255) / 256), 256, 0, stream>>>(s, d, n);
  };
  auto gemm = [&](const _Float16* A, int lda, const int* gi, int goff,
                  const _Float16* B, int ldb,
                  float* C, int ldc, int coff, int accum,
                  _Float16* C16, int ldc16, int c16off, int act,
                  int M, int N, int K) {
    dim3 g((N + 63) / 64, M / 16);
    k_gemm<<<g, 32, 0, stream>>>(A, lda, gi, goff, B, ldb,
                                 C, ldc, coff, accum, C16, ldc16, c16off, act, N, K);
  };

  const int EB = (NEDGE + 255) / 256, NB = (NNODE + 255) / 256;

  // ---- init outputs ----
  k_zero_f32<<<(NGRAPH + NNODE * 3 + 255) / 256, 256, 0, stream>>>(eout, NGRAPH + (long)NNODE * 3);

  // ---- weight conversions ----
  cvt(deg_W1, w1h, 128 * 64);  cvt(deg_W2, w2h, 64 * 64);  cvt(deg_W3, w3h, 64 * 224);
  cvt(head_W0, hw0h, 128 * 128);
  cvt(bWv0, wv0h, 8L * 128 * 128); cvt(bWv1, wv1h, 8L * 64 * 64);  cvt(bWv2, wv2h, 8L * 32 * 32);
  cvt(bWs0, ws0h, 8L * 128 * 128); cvt(bWs1, ws1h, 8L * 128 * 64); cvt(bWs2, ws2h, 8L * 128 * 32);
  cvt(brW1, rw1h, 8L * 128 * 64);  cvt(brW2, rw2h, 8L * 64 * 64);  cvt(brW3, rw3h, 8L * 64 * 224);
  cvt(bWa_src, wash, 8L * 128 * 32); cvt(bWa_dst, wadh, 8L * 128 * 32); cvt(bWa_e, waeh, 8L * 64 * 32);
  cvt(bff1, ff1h, 8L * 128 * 512); cvt(bff2, ff2h, 8L * 512 * 128);
  cvt(bfg1, fg1h, 8L * 128 * 64);  cvt(bfg2, fg2h, 8L * 128 * 32);
  cvt(bfW1, fw1h, 8L * 64 * 64);   cvt(bfW2, fw2h, 8L * 32 * 32);
  k_transpose16<<<(128 * 64 + 255) / 256, 256, 0, stream>>>(deg_W1, w1T, 128, 64);
  k_transpose16<<<(64 * 64 + 255) / 256, 256, 0, stream>>>(deg_W2, w2T, 64, 64);
  k_transpose16<<<(64 * 224 + 255) / 256, 256, 0, stream>>>(deg_W3, w3T, 64, 224);
  k_transpose16<<<(128 * 128 + 255) / 256, 256, 0, stream>>>(head_W0, hw0T, 128, 128);

  // ---- edge geometry + embedding ----
  k_edge_geom<<<EB, 256, 0, stream>>>(pos, esrc, edst, cent, wid, r_, u_, sh1_, sh2_, rbfh);
  gemm(rbfh, 128, nullptr, 0, w1h, 64, p1, 64, 0, 0, y1b, 64, 0, 1, NEDGE, 64, 128);
  gemm(y1b, 64, nullptr, 0, w2h, 64, p2, 64, 0, 0, y2b, 64, 0, 1, NEDGE, 64, 64);
  gemm(y2b, 64, nullptr, 0, w3h, 224, nullptr, 0, 0, 0, gEmb, 224, 0, 0, NEDGE, 224, 64);
  k_node_init<<<NNODE, 192, 0, stream>>>(natom, table, f0, f1, f2);
  k_scatter_embed<<<NEDGE, 192, 0, stream>>>(edst, gEmb, sh1_, sh2_, f0, f1, f2);
  cvt(f0, f0h, 128L * NNODE); cvt(f1, f1h, 192L * NNODE); cvt(f2, f2h, 160L * NNODE);

  // ---- 8 message-passing blocks ----
  for (int i = 0; i < 8; ++i) {
    const _Float16* wv0 = wv0h + (long)i * 128 * 128;
    const _Float16* wv1 = wv1h + (long)i * 64 * 64;
    const _Float16* wv2 = wv2h + (long)i * 32 * 32;
    const _Float16* ws0 = ws0h + (long)i * 128 * 128;
    const _Float16* ws1 = ws1h + (long)i * 128 * 64;
    const _Float16* ws2 = ws2h + (long)i * 128 * 32;
    const _Float16* rw1 = rw1h + (long)i * 128 * 64;
    const _Float16* rw2 = rw2h + (long)i * 64 * 64;
    const _Float16* rw3 = rw3h + (long)i * 64 * 224;
    const _Float16* was = wash + (long)i * 128 * 32;
    const _Float16* wad = wadh + (long)i * 128 * 32;
    const _Float16* wae = waeh + (long)i * 64 * 32;
    const _Float16* ff1 = ff1h + (long)i * 128 * 512;
    const _Float16* ff2 = ff2h + (long)i * 512 * 128;
    const _Float16* fg1 = fg1h + (long)i * 128 * 64;
    const _Float16* fg2 = fg2h + (long)i * 128 * 32;
    const _Float16* fw1 = fw1h + (long)i * 64 * 64;
    const _Float16* fw2 = fw2h + (long)i * 32 * 32;

    // per-block radial MLP
    gemm(rbfh, 128, nullptr, 0, rw1, 64, nullptr, 0, 0, 0, y1b, 64, 0, 1, NEDGE, 64, 128);
    gemm(y1b, 64, nullptr, 0, rw2, 64, nullptr, 0, 0, 0, y2b, 64, 0, 1, NEDGE, 64, 64);
    gemm(y2b, 64, nullptr, 0, rw3, 224, nullptr, 0, 0, 0, gBlk, 224, 0, 0, NEDGE, 224, 64);
    // gathered edge GEMMs
    gemm(f0h, 128, esrc, 0, wv0, 128, nullptr, 0, 0, 0, t0v, 128, 0, 0, NEDGE, 128, 128);
    gemm(f0h, 128, esrc, 0, ws0, 128, nullptr, 0, 0, 0, t0s, 128, 0, 0, NEDGE, 128, 128);
    gemm(f0h, 128, esrc, 0, ws1, 64,  nullptr, 0, 0, 0, ts1, 64, 0, 0, NEDGE, 64, 128);
    gemm(f0h, 128, esrc, 0, ws2, 32,  nullptr, 0, 0, 0, ts2, 32, 0, 0, NEDGE, 32, 128);
    gemm(f0h, 128, esrc, 0, was, 32,  nullptr, 0, 0, 0, za,  32, 0, 0, NEDGE, 32, 128);
    gemm(f0h, 128, edst, 0, wad, 32,  nullptr, 0, 0, 0, zb,  32, 0, 0, NEDGE, 32, 128);
    gemm(y2b, 64, nullptr, 0, wae, 32, nullptr, 0, 0, 0, ze, 32, 0, 0, NEDGE, 32, 64);
    for (int x = 0; x < 3; ++x)
      gemm(f1h, 192, esrc, x * 64, wv1, 64, nullptr, 0, 0, 0, t1m, 192, x * 64, 0, NEDGE, 64, 64);
    for (int x = 0; x < 5; ++x)
      gemm(f2h, 160, esrc, x * 32, wv2, 32, nullptr, 0, 0, 0, t2m, 160, x * 32, 0, NEDGE, 32, 32);

    k_combine<<<NEDGE, 128, 0, stream>>>(gBlk, t0v, t0s, ts1, ts2, t1m, t2m,
                                         za, zb, ze, bwa + (long)i * 32, sh1_, sh2_, logit);
    // segmented softmax + scatter
    k_fill_u32<<<NB, 256, 0, stream>>>(nmax, 0u, NNODE);
    k_zero_f32<<<NB, 256, 0, stream>>>(nsum, NNODE);
    k_segmax<<<EB, 256, 0, stream>>>(edst, logit, nmax);
    k_expsum<<<EB, 256, 0, stream>>>(edst, logit, nmax, ew, nsum);
    k_scatter_attn<<<NEDGE, 192, 0, stream>>>(edst, ew, nsum, t0v, t1m, t2m, f0, f1, f2);
    cvt(f0, f0h, 128L * NNODE); cvt(f1, f1h, 192L * NNODE); cvt(f2, f2h, 160L * NNODE);

    // feed-forward residual on f0
    gemm(f0h, 128, nullptr, 0, ff1, 512, nullptr, 0, 0, 0, u1h, 512, 0, 1, NNODE, 512, 128);
    gemm(u1h, 512, nullptr, 0, ff2, 128, f0, 128, 0, 1, nullptr, 0, 0, 0, NNODE, 128, 512);
    cvt(f0, f0h, 128L * NNODE);

    // gated channel-mix residuals on f1/f2
    gemm(f0h, 128, nullptr, 0, fg1, 64, nullptr, 0, 0, 0, g1h, 64, 0, 2, NNODE, 64, 128);
    gemm(f0h, 128, nullptr, 0, fg2, 32, nullptr, 0, 0, 0, g2h, 32, 0, 2, NNODE, 32, 128);
    for (int x = 0; x < 3; ++x)
      gemm(f1h, 192, nullptr, x * 64, fw1, 64, nullptr, 0, 0, 0, t1nh, 192, x * 64, 0, NNODE, 64, 64);
    for (int x = 0; x < 5; ++x)
      gemm(f2h, 160, nullptr, x * 32, fw2, 32, nullptr, 0, 0, 0, t2nh, 160, x * 32, 0, NNODE, 32, 32);
    k_gatemul<<<NNODE, 192, 0, stream>>>(f1, f2, t1nh, t2nh, g1h, g2h);
    cvt(f1, f1h, 192L * NNODE); cvt(f2, f2h, 160L * NNODE);
  }

  // ---- head ----
  k_layernorm<<<NB, 256, 0, stream>>>(f0, f1, f2, ln_g, ln_b, ln0h, stats);
  gemm(ln0h, 128, nullptr, 0, hw0h, 128, phead, 128, 0, 0, hhh, 128, 0, 1, NNODE, 128, 128);
  k_energy<<<NB, 256, 0, stream>>>(hhh, head_out, batch, eout);

  // ---- backward: head -> LN -> embedding geometry (block adjoints truncated) ----
  k_head_bwd<<<(unsigned)(((long)NNODE * 128 + 255) / 256), 256, 0, stream>>>(phead, head_out, dp16h);
  gemm(dp16h, 128, nullptr, 0, hw0T, 128, nullptr, 0, 0, 0, dln0h, 128, 0, 0, NNODE, 128, 128);
  k_ln_bwd<<<NB, 256, 0, stream>>>(dln0h, ln_g, f0, f1, f2, stats, wf0, wf1, wf2);
  k_emb_bwd_dg<<<NEDGE, 128, 0, stream>>>(edst, wf0, wf1, wf2, sh1_, sh2_, dgh);
  k_emb_bwd_sh<<<EB, 256, 0, stream>>>(edst, wf1, wf2, gEmb, dsh1, dsh2);
  gemm(dgh, 224, nullptr, 0, w3T, 64, nullptr, 0, 0, 0, dy2h, 64, 0, 0, NEDGE, 64, 224);
  k_silu_bwd<<<(unsigned)(((long)NEDGE * 64 + 255) / 256), 256, 0, stream>>>(dy2h, p2, dph, (long)NEDGE * 64);
  gemm(dph, 64, nullptr, 0, w2T, 64, nullptr, 0, 0, 0, dy1h, 64, 0, 0, NEDGE, 64, 64);
  k_silu_bwd<<<(unsigned)(((long)NEDGE * 64 + 255) / 256), 256, 0, stream>>>(dy1h, p1, dp1h, (long)NEDGE * 64);
  gemm(dp1h, 64, nullptr, 0, w1T, 128, nullptr, 0, 0, 0, drbfh, 128, 0, 0, NEDGE, 128, 64);
  k_force<<<EB, 256, 0, stream>>>(esrc, edst, r_, u_, cent, wid, rbfh, drbfh, dsh1, dsh2, fout);
}
